// GConv_13245679140923
// MI455X (gfx1250) — matrix operand order repaired and checked by reading, not compiled, not run on hardware
//
#include <hip/hip_runtime.h>

#define N_NODES 50000
#define N_EDGES 800000
#define D_NODE  192
#define D_EDGE  64
#define D_IN    256
#define D_OUT   256
#define M_TILES 3125          // 50000 / 16
#define N_GROUPS 4            // 4 groups of 64 cols (4x 16-wide WMMA tiles)
#define GEMM_JOBS (M_TILES * N_GROUPS)

typedef __attribute__((ext_vector_type(16))) _Float16 v16h;
typedef __attribute__((ext_vector_type(8)))  float    v8f;

__device__ __forceinline__ void atomAddF(float* p, float v) {
    // lowers to global_atomic_add_f32 (no CAS loop)
    unsafeAtomicAdd(p, v);
}

// ---------------- zero workspace (h + out_deg + in_deg, contiguous) ----------------
__global__ void zero_ws_kernel(float* __restrict__ p, int n) {
    int i = blockIdx.x * blockDim.x + threadIdx.x;
    if (i < n) p[i] = 0.0f;
}

// ---------------- degree histograms ----------------
__global__ void degree_kernel(const int* __restrict__ src, const int* __restrict__ dst,
                              float* __restrict__ outdeg, float* __restrict__ indeg) {
    int e = blockIdx.x * blockDim.x + threadIdx.x;
    if (e < N_EDGES) {
        atomAddF(&outdeg[src[e]], 1.0f);
        atomAddF(&indeg[dst[e]], 1.0f);
    }
}

// ---------------- pre-swizzle weight into per-lane f16 B fragments ----------------
// wsB[((ks*16 + ntile)*32 + lane)*16 + i] = (f16) W[k, n]
//   n = ntile*16 + (lane & 15)
//   k = ks*32 + (lane >= 16 ? 8 : 0) + (i & 7) + (i >= 8 ? 16 : 0)     (ISA 16-bit B 32x16 layout)
__global__ void prep_weight_kernel(const float* __restrict__ w, _Float16* __restrict__ wsB) {
    int t = blockIdx.x * blockDim.x + threadIdx.x;   // 65536 threads
    int i     = t & 15;
    int lane  = (t >> 4) & 31;
    int ntile = (t >> 9) & 15;
    int ks    = (t >> 13) & 7;
    int k = ks * 32 + ((lane >> 4) << 3) + (i & 7) + ((i >> 3) << 4);
    int n = ntile * 16 + (lane & 15);
    wsB[t] = (_Float16)w[k * D_OUT + n];
}

// ---------------- edge aggregation: h[dst] += concat(feat_n[src], edge_feat) ----------------
// one thread per (edge, 4-float chunk); 64 chunks/edge (48 node-feat + 16 edge-feat)
__global__ void aggregate_kernel(const float* __restrict__ feat,
                                 const float* __restrict__ efeat,
                                 const int* __restrict__ src,
                                 const int* __restrict__ dst,
                                 const float* __restrict__ outdeg,
                                 float* __restrict__ h) {
    long long idx = (long long)blockIdx.x * blockDim.x + threadIdx.x;
    if (idx >= (long long)N_EDGES * 64) return;
    int e = (int)(idx >> 6);
    int c = (int)(idx & 63);
    int d = dst[e];
    float* hrow = h + (size_t)d * D_IN;
    if (c < 48) {
        int s = src[e];
        float sc = rsqrtf(fmaxf(outdeg[s], 1.0f));
        const float4 v = *(const float4*)(feat + (size_t)s * D_NODE + c * 4);
        int o = c * 4;
        atomAddF(hrow + o + 0, v.x * sc);
        atomAddF(hrow + o + 1, v.y * sc);
        atomAddF(hrow + o + 2, v.z * sc);
        atomAddF(hrow + o + 3, v.w * sc);
    } else {
        int ce = c - 48;
        const float4 v = *(const float4*)(efeat + (size_t)e * D_EDGE + ce * 4);
        int o = D_NODE + ce * 4;
        atomAddF(hrow + o + 0, v.x);
        atomAddF(hrow + o + 1, v.y);
        atomAddF(hrow + o + 2, v.z);
        atomAddF(hrow + o + 3, v.w);
    }
}

// ---------------- GEMM: out = (h @ W) * indeg^-1/2 + bias, via v_wmma_f32_16x16x32_f16 ----
// one wave -> 16x64 output tile (4 WMMA N-tiles, A fragment reused), K = 256 = 8 steps
__global__ void __launch_bounds__(256)
gemm_wmma_kernel(const float* __restrict__ h, const _Float16* __restrict__ wsB,
                 const float* __restrict__ indeg, const float* __restrict__ bias,
                 float* __restrict__ out) {
    int wave = threadIdx.x >> 5;
    int lane = threadIdx.x & 31;
    int job = blockIdx.x * 8 + wave;
    if (job >= GEMM_JOBS) return;            // wave-granular: EXEC all-ones inside
    int mtile = job >> 2;
    int ng    = job & 3;
    int m0 = mtile * 16;

    v8f c0 = {}, c1 = {}, c2 = {}, c3 = {};

    // A fragment base: row = m0 + (lane&15); half-wave k-offset 0 or 8 (ISA 16-bit A layout)
    const float* abase = h + (size_t)(m0 + (lane & 15)) * D_IN + ((lane >> 4) << 3);
    // B fragments: contiguous v16h per lane from pre-swizzled weight
    const v16h* bbase = (const v16h*)wsB + (ng * 4) * 32 + lane;

#pragma unroll
    for (int ks = 0; ks < 8; ++ks) {
        const float* ar = abase + ks * 32;
        v16h a;
#pragma unroll
        for (int i = 0; i < 8; ++i) {
            a[i]     = (_Float16)ar[i];        // K = kbase + 0..7
            a[i + 8] = (_Float16)ar[i + 16];   // K = kbase + 16..23
        }
        const v16h* bp = bbase + ks * 512;     // 16 ntiles * 32 lanes per k-step
        c0 = __builtin_amdgcn_wmma_f32_16x16x32_f16(false, a, false, bp[0],  (short)0, c0, false, false);
        c1 = __builtin_amdgcn_wmma_f32_16x16x32_f16(false, a, false, bp[32], (short)0, c1, false, false);
        c2 = __builtin_amdgcn_wmma_f32_16x16x32_f16(false, a, false, bp[64], (short)0, c2, false, false);
        c3 = __builtin_amdgcn_wmma_f32_16x16x32_f16(false, a, false, bp[96], (short)0, c3, false, false);
    }

    // epilogue: C/D layout -> VGPR j holds row j (lanes 0-15) or row j+8 (lanes 16-31)
    int col = lane & 15;
    int nb = ng * 64 + col;
    float b0 = bias[nb], b1 = bias[nb + 16], b2 = bias[nb + 32], b3 = bias[nb + 48];
#pragma unroll
    for (int j = 0; j < 8; ++j) {
        int row = (lane < 16) ? j : (j + 8);
        int gm = m0 + row;
        float sc = rsqrtf(fmaxf(indeg[gm], 1.0f));
        float* orow = out + (size_t)gm * D_OUT + ng * 64 + col;
        orow[0]  = c0[j] * sc + b0;
        orow[16] = c1[j] * sc + b1;
        orow[32] = c2[j] * sc + b2;
        orow[48] = c3[j] * sc + b3;
    }
}

extern "C" void kernel_launch(void* const* d_in, const int* in_sizes, int n_in,
                              void* d_out, int out_size, void* d_ws, size_t ws_size,
                              hipStream_t stream) {
    (void)in_sizes; (void)n_in; (void)out_size; (void)ws_size;
    const float* feat   = (const float*)d_in[0];
    const float* efeat  = (const float*)d_in[1];
    const int*   src    = (const int*)d_in[2];
    const int*   dst    = (const int*)d_in[3];
    const float* weight = (const float*)d_in[4];
    const float* bias   = (const float*)d_in[5];
    float* out = (float*)d_out;

    // workspace layout: h[N,256] | outdeg[N] | indeg[N] | wsB (65536 f16)
    float* h      = (float*)d_ws;
    float* outdeg = h + (size_t)N_NODES * D_IN;
    float* indeg  = outdeg + N_NODES;
    _Float16* wsB = (_Float16*)(indeg + N_NODES);   // 32B-aligned (offset 51.6MB)

    int zn = N_NODES * D_IN + 2 * N_NODES;          // zero h + both degree arrays
    zero_ws_kernel<<<(zn + 255) / 256, 256, 0, stream>>>(h, zn);

    prep_weight_kernel<<<65536 / 256, 256, 0, stream>>>(weight, wsB);

    degree_kernel<<<(N_EDGES + 255) / 256, 256, 0, stream>>>(src, dst, outdeg, indeg);

    long long agg_items = (long long)N_EDGES * 64;
    aggregate_kernel<<<(unsigned)((agg_items + 255) / 256), 256, 0, stream>>>(
        feat, efeat, src, dst, outdeg, h);

    gemm_wmma_kernel<<<(GEMM_JOBS + 7) / 8, 256, 0, stream>>>(h, wsB, indeg, bias, out);
}
